// GAT_69810398429280
// MI455X (gfx1250) — matrix operand order repaired and checked by reading, not compiled, not run on hardware
//
#include <hip/hip_runtime.h>
#include <hip/hip_bf16.h>

// ---- problem constants (from reference) ------------------------------------
#define B_   8
#define C_   64
#define N_   4096
#define MTILE 128          // rows of i per workgroup (8 waves x 16)
#define KT    32           // j-tile (K of wmma f16 16x16x32)
#define NWAVE 8
#define XS_STRIDE 40       // halves per c-row in LDS x tile (pad vs 32)

typedef __attribute__((ext_vector_type(16))) _Float16 v16h;
typedef __attribute__((ext_vector_type(8)))  float    v8f;
typedef __attribute__((ext_vector_type(4)))  unsigned u32x4;
typedef __attribute__((ext_vector_type(4)))  float    f32x4;

union AFrag { v16h v; _Float16 h[16]; u32x4 q[2]; };
union Acc   { v8f  v; float f[8]; };
union H8    { _Float16 h[8]; u32x4 q; };

#define LOG2E 1.4426950408889634f

__device__ __forceinline__ float leaky01(float s) {
    return fmaxf(s, 0.0f) + 0.01f * fminf(s, 0.0f);
}

// ---- kernel 0: reset per-batch max accumulator -----------------------------
__global__ void gat_init(unsigned* __restrict__ fmaxbits) {
    if (threadIdx.x < B_) fmaxbits[threadIdx.x] = 0u;   // ordered-uint identity
}

// ---- kernel 1: f[b,n] = sum_c w[c]*x[b,c,n]; fmax[b] via ordered-uint max --
__global__ __launch_bounds__(256) void gat_f(const float* __restrict__ x,
                                             const float* __restrict__ w,
                                             float* __restrict__ fbuf,
                                             unsigned* __restrict__ fmaxbits) {
    const int tid = threadIdx.x;
    const int b   = blockIdx.x >> 4;                    // 16 blocks per batch
    const int n   = ((blockIdx.x & 15) << 8) + tid;
    const float* xb = x + (size_t)b * C_ * N_ + n;
    float acc = 0.0f;
#pragma unroll 8
    for (int c = 0; c < C_; ++c) acc += w[c] * xb[(size_t)c * N_];
    fbuf[b * N_ + n] = acc;

    // monotone float->uint encoding so unsigned max == float max
    unsigned u = __float_as_uint(acc);
    u = (u & 0x80000000u) ? ~u : (u | 0x80000000u);
#pragma unroll
    for (int s = 16; s > 0; s >>= 1) {
        unsigned o = (unsigned)__shfl_xor((int)u, s, 32);
        u = (o > u) ? o : u;
    }
    if ((tid & 31) == 0) atomicMax(&fmaxbits[b], u);
}

// ---- kernel 2: fused softmax(leaky(f_i+f_j)) @ x^T, then ELU ---------------
__global__ __launch_bounds__(256) void gat_main(const float* __restrict__ x,
                                                const float* __restrict__ fbuf,
                                                const unsigned* __restrict__ fmaxbits,
                                                float* __restrict__ out) {
    __shared__ float    fsh[N_];                        // 16 KB: f[b,:]
    __shared__ _Float16 xs[C_ * XS_STRIDE];             // 5 KB: x tile (c-major)
    __shared__ float    outsh[NWAVE * C_ * 17];         // 34 KB: store transpose

    const int tid    = threadIdx.x;
    const int b      = blockIdx.x >> 5;                 // 32 row-blocks / batch
    const int rowblk = blockIdx.x & 31;
    const int i0b    = rowblk * MTILE;
    const int wave   = tid >> 5;
    const int lane   = tid & 31;
    const int m      = lane & 15;
    const int hi     = lane >> 4;                       // 0: lanes 0-15, 1: 16-31

    const float* xb = x + (size_t)b * C_ * N_;

    // stage f[b,:] into LDS
    const float* fb = fbuf + b * N_;
    for (int t = tid; t < N_; t += 256) fsh[t] = fb[t];
    __syncthreads();

    // decode fmax[b]
    unsigned u  = fmaxbits[b];
    unsigned db = (u & 0x80000000u) ? (u & 0x7fffffffu) : ~u;
    const float fmax = __uint_as_float(db);

    // this lane's row and its closed-form softmax max: m_i = leaky(f_i + fmax)
    const int   irow = i0b + wave * 16 + m;
    const float fi   = fsh[irow];
    const float mi   = leaky01(fi + fmax);

    Acc acc[4];
#pragma unroll
    for (int q = 0; q < 4; ++q)
#pragma unroll
        for (int r = 0; r < 8; ++r) acc[q].f[r] = 0.0f;

    float zsum = 0.0f;
    const int kb_b = hi ? 8 : 0;    // B-frag K base (lanes 16-31 hold K=8..15, 24..31)
    const int kb_a = hi ? 8  : 0;   // A-frag K base within each 16-group

    const int stage_c = tid >> 2;             // staging: c per thread
    const int stage_j = (tid & 3) << 3;       // 8 consecutive j per thread

    for (int jt = 0; jt < N_; jt += KT) {
        __syncthreads();
        // ---- stage x[b, 0..63, jt..jt+31] -> LDS as f16, c-major -----------
        {
            const f32x4* src =
                reinterpret_cast<const f32x4*>(xb + (size_t)stage_c * N_ + jt + stage_j);
            f32x4 p0 = src[0];
            f32x4 p1 = src[1];
            H8 hh;
#pragma unroll
            for (int e = 0; e < 4; ++e) { hh.h[e] = (_Float16)p0[e]; hh.h[4 + e] = (_Float16)p1[e]; }
            *reinterpret_cast<u32x4*>(&xs[stage_c * XS_STRIDE + stage_j]) = hh.q;
            if (jt + KT < N_)   // pull next tile toward L2/WGP$
                __builtin_prefetch(xb + (size_t)stage_c * N_ + jt + KT + stage_j, 0, 0);
        }
        __syncthreads();

        // ---- A fragment: unnormalized softmax probs, ISA 16-bit A layout ---
        // lanes 0-15:  v0-3 -> K=0..7,   v4-7 -> K=16..23  (rows M=0..15)
        // lanes 16-31: v0-3 -> K=8..15,  v4-7 -> K=24..31
        AFrag a;
        float zl = 0.0f;
#pragma unroll
        for (int g = 0; g < 2; ++g) {
#pragma unroll
            for (int vv = 0; vv < 4; ++vv) {
                const int k0  = g * 16 + kb_a + 2 * vv;
                const float s0 = leaky01(fi + fsh[jt + k0]);
                const float s1 = leaky01(fi + fsh[jt + k0 + 1]);
                const float p0 = __builtin_amdgcn_exp2f((s0 - mi) * LOG2E);
                const float p1 = __builtin_amdgcn_exp2f((s1 - mi) * LOG2E);
                zl += p0 + p1;
                a.h[(g * 4 + vv) * 2]     = (_Float16)p0;
                a.h[(g * 4 + vv) * 2 + 1] = (_Float16)p1;
            }
        }
        zsum += zl;

        // ---- B fragments (32j x 16c per chunk) + 4 WMMAs -------------------
#pragma unroll
        for (int q = 0; q < 4; ++q) {
            const int c = q * 16 + m;
            AFrag bf;
            const u32x4* bsrc =
                reinterpret_cast<const u32x4*>(&xs[c * XS_STRIDE + kb_b]);
            bf.q[0] = bsrc[0];
            bf.q[1] = bsrc[2];
            acc[q].v = __builtin_amdgcn_wmma_f32_16x16x32_f16(
                false, a.v, false, bf.v, (short)0, acc[q].v, false, false);
        }
    }

    // ---- row sums: lanes L and L+16 hold disjoint K halves of row m --------
    float ztot = zsum + __shfl_xor(zsum, 16, 32);
    float zinv[8];
#pragma unroll
    for (int r = 0; r < 8; ++r) {
        const int row = (hi << 3) + r;                 // D layout: M = r + 8*hi
        zinv[r] = __builtin_amdgcn_rcpf(__shfl(ztot, row, 32));
    }

    // ---- normalize + ELU, transpose via LDS for coalesced stores -----------
    float* osh = &outsh[wave * (C_ * 17)];
#pragma unroll
    for (int q = 0; q < 4; ++q) {
        const int c = q * 16 + m;
#pragma unroll
        for (int r = 0; r < 8; ++r) {
            float v = acc[q].f[r] * zinv[r];
            const float e = __builtin_amdgcn_exp2f(v * LOG2E) - 1.0f;
            v = v > 0.0f ? v : e;
            osh[c * 17 + (hi << 3) + r] = v;
        }
    }
    __syncthreads();
    {
        const int c   = tid >> 2;
        const int ist = (tid & 3) << 5;                // 32 consecutive i
        float* dst = out + (size_t)b * C_ * N_ + (size_t)c * N_ + i0b + ist;
#pragma unroll
        for (int e = 0; e < 32; e += 4) {
            f32x4 val;
#pragma unroll
            for (int t = 0; t < 4; ++t) {
                const int i1 = ist + e + t;
                val[t] = outsh[(i1 >> 4) * (C_ * 17) + c * 17 + (i1 & 15)];
            }
            *reinterpret_cast<f32x4*>(dst + e) = val;
        }
    }
}

// ---- host launcher ---------------------------------------------------------
extern "C" void kernel_launch(void* const* d_in, const int* in_sizes, int n_in,
                              void* d_out, int out_size, void* d_ws, size_t ws_size,
                              hipStream_t stream) {
    (void)in_sizes; (void)n_in; (void)out_size; (void)ws_size;
    const float* x = (const float*)d_in[0];   // (B, C, N) f32
    const float* w = (const float*)d_in[1];   // (C,)      f32
    float* out = (float*)d_out;               // (B, C, N) f32

    float*    fbuf     = (float*)d_ws;                                  // B*N f32
    unsigned* fmaxbits = (unsigned*)((char*)d_ws + (size_t)B_ * N_ * sizeof(float));

    gat_init<<<1, 32, 0, stream>>>(fmaxbits);
    gat_f<<<B_ * (N_ / 256), 256, 0, stream>>>(x, w, fbuf, fmaxbits);
    gat_main<<<B_ * (N_ / MTILE), 256, 0, stream>>>(x, fbuf, fmaxbits, out);
}
